// LoGA_28363964022814
// MI455X (gfx1250) — matrix operand order repaired and checked by reading, not compiled, not run on hardware
//
#include <hip/hip_runtime.h>
#include <hip/hip_bf16.h>

#define BB 32
#define SS 4096
#define CIN 32
#define DD 256
#define HH 8
#define PP 24
#define WW 64
#define DFFN 512
#define KK 5
#define DHH 32
#define MM 127
#define MTILES 8
#define EPSV 1e-5f
#define SMSCALE 0.17677669529663687f  /* 1/sqrt(32) */
#define SROW (SS + 4)                 /* padded LDS score row stride */

typedef unsigned short u16;
typedef __attribute__((ext_vector_type(16))) __bf16 v16bf;
typedef __attribute__((ext_vector_type(8)))  float  v8f;
typedef __attribute__((ext_vector_type(8)))  unsigned int v8u;

static __device__ __forceinline__ u16 f2bf(float f) {
  unsigned u = __float_as_uint(f);
  unsigned r = u + 0x7FFFu + ((u >> 16) & 1u);   // round-to-nearest-even
  return (u16)(r >> 16);
}
static __device__ __forceinline__ float bf2f(u16 h) {
  return __uint_as_float(((unsigned)h) << 16);
}
static __device__ __forceinline__ unsigned pk2(u16 lo, u16 hi) {
  return (unsigned)lo | ((unsigned)hi << 16);
}
// A-matrix (16-bit, 16x32) packed-pair K index for VGPR j, per ISA layout
static __device__ __forceinline__ int klA(int lane, int j) {
  return ((lane >> 4) << 3) + ((j >> 2) << 4) + ((j & 3) << 1);
}
// B-matrix (16-bit, 32x16) packed-pair K index for VGPR v
static __device__ __forceinline__ int klB(int lane, int v) {
  return ((lane >> 4) << 3) + ((v >> 2) << 4) + ((v & 3) << 1);
}
// A fragment from a contiguous 32-element bf16 K-slice: two b128 loads
static __device__ __forceinline__ v8u ldfragA(const u16* __restrict__ p, int lane) {
  const u16* q = p + ((lane & 16) ? 8 : 0);
  uint4 lo = *(const uint4*)(q);
  uint4 hi = *(const uint4*)(q + 16);
  v8u r;
  r[0] = lo.x; r[1] = lo.y; r[2] = lo.z; r[3] = lo.w;
  r[4] = hi.x; r[5] = hi.y; r[6] = hi.z; r[7] = hi.w;
  return r;
}
// B fragment from a contiguous 32-element bf16 K-slice: two b128 loads
static __device__ __forceinline__ v8u ldfragB(const u16* __restrict__ p, int lane) {
  const u16* q = p + ((lane & 16) ? 8 : 0);
  uint4 lo = *(const uint4*)(q);
  uint4 hi = *(const uint4*)(q + 16);
  v8u r;
  r[0] = lo.x; r[1] = lo.y; r[2] = lo.z; r[3] = lo.w;
  r[4] = hi.x; r[5] = hi.y; r[6] = hi.z; r[7] = hi.w;
  return r;
}
static __device__ __forceinline__ float gelu_f(float x) {
  return 0.5f * x * (1.0f + erff(x * 0.7071067811865475f));
}
static __device__ __forceinline__ v8f wmma_bf16(v8u a, v8u b, v8f c) {
#if __HIP_DEVICE_COMPILE__
  return __builtin_amdgcn_wmma_f32_16x16x32_bf16(
      false, __builtin_bit_cast(v16bf, a),
      false, __builtin_bit_cast(v16bf, b),
      (short)0, c, false, false);
#else
  (void)a; (void)b;
  return c;
#endif
}
// Row-wise LayerNorm over n values held one-per-thread; red is shared[n]
static __device__ __forceinline__ float block_ln(float acc, float* red, int tid, int n) {
  red[tid] = acc; __syncthreads();
  for (int s = n >> 1; s > 0; s >>= 1) { if (tid < s) red[tid] += red[tid + s]; __syncthreads(); }
  float mean = red[0] / (float)n; __syncthreads();
  float dv = acc - mean;
  red[tid] = dv * dv; __syncthreads();
  for (int s = n >> 1; s > 0; s >>= 1) { if (tid < s) red[tid] += red[tid + s]; __syncthreads(); }
  float var = red[0] / (float)n; __syncthreads();
  return dv * rsqrtf(var + EPSV);
}

__global__ void k_zero(float* p, int n) {
  int i = blockIdx.x * blockDim.x + threadIdx.x;
  if (i < n) p[i] = 0.f;
}

// ---- one-time fp32 -> bf16 weight conversion ----
__global__ void k_cvtw(const float* __restrict__ w, u16* __restrict__ o, int n) {
  int i = blockIdx.x * blockDim.x + threadIdx.x;
  if (i < n) o[i] = f2bf(w[i]);
}

// ---- depthwise conv + pointwise (WMMA, K=32) + GELU + BN stat accumulation ----
__global__ void k_frontend(const float* __restrict__ x, const float* __restrict__ dww,
                           const u16* __restrict__ pwb, u16* __restrict__ z,
                           float* __restrict__ stats) {
  __shared__ float conv[64][CIN];
  __shared__ float lsum[DD], lsq[DD];
  const int tid = threadIdx.x;
  const int blk = blockIdx.x;
  const int b = blk / (SS / 64);
  const int s0 = (blk % (SS / 64)) * 64;
  lsum[tid] = 0.f; lsq[tid] = 0.f;
  for (int i = tid; i < 64 * CIN; i += 256) {
    int sr = i >> 5, c = i & 31;
    int s = s0 + sr;
    float acc = 0.f;
#pragma unroll
    for (int k = 0; k < KK; ++k) {
      int ss = s + k - 2;
      if (ss >= 0 && ss < SS) acc += x[((size_t)b * SS + ss) * CIN + c] * dww[c * KK + k];
    }
    conv[sr][c] = acc;
  }
  __syncthreads();
  const int wid = tid >> 5, lane = tid & 31;
  const int mcol = lane & 15;
  for (int i = 0; i < 8; ++i) {
    int t = wid * 8 + i;             // 64 tiles = 4 row-tiles x 16 col-tiles
    int rt = t >> 4, ct = t & 15;
    v8u a;
    int row = rt * 16 + mcol;
#pragma unroll
    for (int j = 0; j < 8; ++j) {
      int k0 = klA(lane, j);
      a[j] = pk2(f2bf(conv[row][k0]), f2bf(conv[row][k0 + 1]));
    }
    int d = ct * 16 + mcol;
    v8u bm = ldfragB(pwb + d * CIN, lane);   // CIN==32: one K-slice
    v8f c = {};
    c = wmma_bf16(a, bm, c);
    float ps = 0.f, pq = 0.f;
#pragma unroll
    for (int r = 0; r < 8; ++r) {
      int rr = r + ((lane >> 4) << 3);
      float g = gelu_f(c[r]);
      z[((size_t)b * SS + s0 + rt * 16 + rr) * DD + d] = f2bf(g);
      ps += g; pq += g * g;
    }
    atomicAdd(&lsum[d], ps);
    atomicAdd(&lsq[d], pq);
  }
  __syncthreads();
  atomicAdd(&stats[tid], lsum[tid]);
  atomicAdd(&stats[DD + tid], lsq[tid]);
}

__global__ void k_bnfin(float* stats, const float* g, const float* bch) {
  int d = threadIdx.x;
  const float n = (float)BB * (float)SS;
  float mean = stats[d] / n;
  float var = stats[DD + d] / n - mean * mean;
  float sc = g[d] * rsqrtf(var + EPSV);
  stats[2 * DD + d] = sc;
  stats[3 * DD + d] = bch[d] - mean * sc;
}

__global__ void k_bnapply(u16* z, const float* __restrict__ stats) {
  int tid = threadIdx.x;
  float sc = stats[2 * DD + tid], bi = stats[3 * DD + tid];
  size_t base = (size_t)blockIdx.x * 2048 + tid;
#pragma unroll
  for (int j = 0; j < 8; ++j) {
    size_t idx = base + (size_t)j * 256;
    z[idx] = f2bf(bf2f(z[idx]) * sc + bi);
  }
}

__global__ void k_qmean(const u16* __restrict__ z, float* __restrict__ Qf,
                        u16* __restrict__ Qb) {
  int blk = blockIdx.x;
  int b = blk / MM, m = blk % MM;
  int d = threadIdx.x;
  const u16* p = z + ((size_t)b * SS + 32 * m) * DD + d;
  float acc = 0.f;
  for (int w = 0; w < WW; ++w) acc += bf2f(p[(size_t)w * DD]);
  acc *= (1.0f / (float)WW);
  Qf[(size_t)blk * DD + d] = acc;
  Qb[(size_t)blk * DD + d] = f2bf(acc);
}

// ---- generic bf16 WMMA GEMM: out[R,256] = A[R,256] @ Wb^T, Wb bf16 row-major [256,256] ----
__global__ void k_proj(const u16* __restrict__ A, const u16* __restrict__ Wb,
                       u16* __restrict__ out) {
  const int tid = threadIdx.x;
  const int wid = tid >> 5, lane = tid & 31;
  const int gw = blockIdx.x * 8 + wid;
  const int rt = gw >> 4, ct = gw & 15;
  const int mcol = lane & 15;
  const size_t arow = (size_t)(rt * 16 + mcol) * DD;
  const int n = ct * 16 + mcol;
  v8f c = {};
#pragma unroll
  for (int kk = 0; kk < 8; ++kk) {
    v8u a  = ldfragA(A + arow + kk * 32, lane);
    v8u bm = ldfragB(Wb + (size_t)n * DD + kk * 32, lane);
    c = wmma_bf16(a, bm, c);
  }
#pragma unroll
  for (int r = 0; r < 8; ++r) {
    int row = rt * 16 + r + ((lane >> 4) << 3);
    out[(size_t)row * DD + ct * 16 + mcol] = f2bf(c[r]);
  }
}

// ---- attention: QK^T (WMMA) -> 256KB LDS scores -> softmax -> attn_g + PV (WMMA) ----
__global__ void k_attn(const u16* __restrict__ Qh, const u16* __restrict__ Kp,
                       const u16* __restrict__ Vp, float* __restrict__ attng,
                       float* __restrict__ Obuf) {
  extern __shared__ float sm[];
  float* sc   = sm;                  // 16 * SROW scores
  float* red  = sm + 16 * SROW;      // 256
  float* rst  = red + 256;           // 32 (row max / inv-sum)
  float* olds = rst + 32;            // 16*32 O accumulator
  const int tid = threadIdx.x;
  const int wid = tid >> 5, lane = tid & 31;
  const int mt = blockIdx.x % MTILES;
  const int bh = blockIdx.x / MTILES;
  const int h = bh % HH, b = bh / HH;
  const int mcol = lane & 15;
  // Q fragment (16 queries x DH=32, one WMMA K step)
  v8u aQ;
  {
    int q = mt * 16 + mcol;
    const u16* qp = Qh + (size_t)(b * MM + (q < MM ? q : 0)) * DD + h * DHH;
    aQ = ldfragA(qp, lane);
    if (q >= MM) { v8u zz = {}; aQ = zz; }
  }
  // scores: each wave covers 512 keys = 32 WMMAs of 16 keys
  const int kbase0 = wid * (SS / 8);
  for (int t = 0; t < (SS / 8) / 16; ++t) {
    int kb = kbase0 + t * 16;
    const u16* kp = Kp + ((size_t)b * SS + kb + mcol) * DD + h * DHH;
    v8u bm = ldfragB(kp, lane);
    v8f c = {};
    c = wmma_bf16(aQ, bm, c);
#pragma unroll
    for (int r = 0; r < 8; ++r) {
      int rr = r + ((lane >> 4) << 3);
      sc[rr * SROW + kb + mcol] = c[r] * SMSCALE;
    }
  }
  __syncthreads();
  // softmax: 16 threads per row over 4096 cols; write fp32 probs to attn_g
  {
    const int row = tid >> 4, t16 = tid & 15;
    float* sr = sc + row * SROW;
    float mx = -3.4e38f;
    for (int cI = t16; cI < SS; cI += 16) mx = fmaxf(mx, sr[cI]);
    red[row * 16 + t16] = mx; __syncthreads();
    if (t16 == 0) {
      float m2 = red[row * 16];
      for (int i = 1; i < 16; ++i) m2 = fmaxf(m2, red[row * 16 + i]);
      rst[row] = m2;
    }
    __syncthreads();
    float rm = rst[row];
    float s = 0.f;
    for (int cI = t16; cI < SS; cI += 16) {
      float e = __expf(sr[cI] - rm);
      sr[cI] = e; s += e;
    }
    red[row * 16 + t16] = s; __syncthreads();
    if (t16 == 0) {
      float s2 = 0.f;
      for (int i = 0; i < 16; ++i) s2 += red[row * 16 + i];
      rst[16 + row] = 1.0f / s2;
    }
    __syncthreads();
    float inv = rst[16 + row];
    int mg = mt * 16 + row;
    float* og = attng + ((size_t)(b * HH + h) * MM + (mg < MM ? mg : 0)) * SS;
    for (int cI = t16; cI < SS; cI += 16) {
      float pv = sr[cI] * inv;
      sr[cI] = pv;
      if (mg < MM) og[cI] = pv;
    }
  }
  for (int i = tid; i < 16 * 32; i += 256) olds[i] = 0.f;
  __syncthreads();
  // PV: each wave accumulates over its key blocks (32 keys = one K step)
  {
    v8f c0 = {}, c1 = {};
    for (int kb8 = wid; kb8 < SS / 32; kb8 += 8) {
      int kb = kb8 * 32;
      v8u aP, b0, b1;
#pragma unroll
      for (int j = 0; j < 8; ++j) {
        int k0 = klA(lane, j);
        const float* pr = sc + mcol * SROW + kb + k0;
        aP[j] = pk2(f2bf(pr[0]), f2bf(pr[1]));
      }
      const u16* vp = Vp + ((size_t)b * SS + kb) * DD + h * DHH;
#pragma unroll
      for (int v = 0; v < 8; ++v) {
        int k0 = klB(lane, v);
        b0[v] = pk2(vp[(size_t)k0 * DD + mcol],      vp[(size_t)(k0 + 1) * DD + mcol]);
        b1[v] = pk2(vp[(size_t)k0 * DD + 16 + mcol], vp[(size_t)(k0 + 1) * DD + 16 + mcol]);
      }
      c0 = wmma_bf16(aP, b0, c0);
      c1 = wmma_bf16(aP, b1, c1);
    }
#pragma unroll
    for (int r = 0; r < 8; ++r) {
      int rr = r + ((lane >> 4) << 3);
      atomicAdd(&olds[rr * 32 + mcol], c0[r]);
      atomicAdd(&olds[rr * 32 + 16 + mcol], c1[r]);
    }
  }
  __syncthreads();
  for (int i = tid; i < 512; i += 256) {
    int rr = i >> 5, dh = i & 31;
    int qg = mt * 16 + rr;
    if (qg < MM) Obuf[(size_t)(b * MM + qg) * DD + h * DHH + dh] = olds[i];
  }
}

// ---- Hh = LN(O @ Wo^T + Q) ----
__global__ void k_lgout(const float* __restrict__ Obuf, const float* __restrict__ Wo,
                        const float* __restrict__ Qf, float* __restrict__ Hh) {
  __shared__ float orow[DD];
  __shared__ float red[DD];
  const int tid = threadIdx.x;
  const size_t row = blockIdx.x;
  orow[tid] = Obuf[row * DD + tid];
  __syncthreads();
  float acc = Qf[row * DD + tid];
  const float* wr = Wo + tid * DD;
  for (int k = 0; k < DD; ++k) acc += orow[k] * wr[k];
  Hh[row * DD + tid] = block_ln(acc, red, tid, DD);
}

__global__ void k_decq(const float* __restrict__ Qd, const float* __restrict__ Wq,
                       float* __restrict__ q2) {
  __shared__ float qr[DD];
  int tid = threadIdx.x, p = blockIdx.x;
  qr[tid] = Qd[p * DD + tid];
  __syncthreads();
  float acc = 0.f;
  const float* wr = Wq + tid * DD;
  for (int k = 0; k < DD; ++k) acc += qr[k] * wr[k];
  q2[p * DD + tid] = acc;
}

__global__ void k_deckv(const float* __restrict__ Hh, const float* __restrict__ Wk,
                        const float* __restrict__ Wv, float* __restrict__ k2,
                        float* __restrict__ v2) {
  __shared__ float hr[DD];
  int tid = threadIdx.x; size_t row = blockIdx.x;
  hr[tid] = Hh[row * DD + tid];
  __syncthreads();
  float ak = 0.f, av = 0.f;
  const float* wk = Wk + tid * DD;
  const float* wv = Wv + tid * DD;
  for (int k = 0; k < DD; ++k) { ak += hr[k] * wk[k]; av += hr[k] * wv[k]; }
  k2[row * DD + tid] = ak;
  v2[row * DD + tid] = av;
}

__global__ void k_decattn(const float* __restrict__ q2, const float* __restrict__ k2,
                          const float* __restrict__ v2, float* __restrict__ attnd,
                          float* __restrict__ o2) {
  __shared__ float qh[PP * DHH];
  __shared__ float kh[MM * DHH];
  __shared__ float vh[MM * DHH];
  __shared__ float scd[PP * MM];
  int tid = threadIdx.x;
  int b = blockIdx.x / HH, h = blockIdx.x % HH;
  for (int i = tid; i < PP * DHH; i += 256) {
    int p = i >> 5, dh = i & 31;
    qh[i] = q2[p * DD + h * DHH + dh];
  }
  for (int i = tid; i < MM * DHH; i += 256) {
    int j = i >> 5, dh = i & 31;
    kh[i] = k2[((size_t)b * MM + j) * DD + h * DHH + dh];
    vh[i] = v2[((size_t)b * MM + j) * DD + h * DHH + dh];
  }
  __syncthreads();
  for (int i = tid; i < PP * MM; i += 256) {
    int p = i / MM, j = i % MM;
    float a = 0.f;
#pragma unroll
    for (int d = 0; d < DHH; ++d) a += qh[p * DHH + d] * kh[j * DHH + d];
    scd[i] = a * SMSCALE;
  }
  __syncthreads();
  if (tid < PP) {
    float* srow = scd + tid * MM;
    float mx = srow[0];
    for (int j = 1; j < MM; ++j) mx = fmaxf(mx, srow[j]);
    float s = 0.f;
    for (int j = 0; j < MM; ++j) { float e = __expf(srow[j] - mx); srow[j] = e; s += e; }
    float inv = 1.0f / s;
    float* og = attnd + ((size_t)(b * HH + h) * PP + tid) * MM;
    for (int j = 0; j < MM; ++j) { float pv = srow[j] * inv; srow[j] = pv; og[j] = pv; }
  }
  __syncthreads();
  for (int i = tid; i < PP * DHH; i += 256) {
    int p = i >> 5, dh = i & 31;
    float a = 0.f;
    for (int j = 0; j < MM; ++j) a += scd[p * MM + j] * vh[j * DHH + dh];
    o2[((size_t)b * PP + p) * DD + h * DHH + dh] = a;
  }
}

__global__ void k_decz2(const float* __restrict__ o2, const float* __restrict__ Wo,
                        const float* __restrict__ Qd, float* __restrict__ z2) {
  __shared__ float orow[DD];
  __shared__ float red[DD];
  int tid = threadIdx.x; size_t row = blockIdx.x;
  int p = (int)(row % PP);
  orow[tid] = o2[row * DD + tid];
  __syncthreads();
  float acc = Qd[p * DD + tid];
  const float* wr = Wo + tid * DD;
  for (int k = 0; k < DD; ++k) acc += orow[k] * wr[k];
  z2[row * DD + tid] = block_ln(acc, red, tid, DD);
}

__global__ void k_ffn1(const float* __restrict__ z2, const float* __restrict__ W1,
                       const float* __restrict__ b1, float* __restrict__ u1) {
  __shared__ float zr[DD];
  int tid = threadIdx.x; size_t row = blockIdx.x;
  zr[tid] = z2[row * DD + tid];
  __syncthreads();
  for (int o = tid; o < DFFN; o += 256) {
    float acc = b1[o];
    const float* wr = W1 + o * DD;
    for (int k = 0; k < DD; ++k) acc += zr[k] * wr[k];
    u1[row * DFFN + o] = gelu_f(acc);
  }
}

__global__ void k_ffn2(const float* __restrict__ u1, const float* __restrict__ W2,
                       const float* __restrict__ b2, const float* __restrict__ z2,
                       float* __restrict__ uu) {
  __shared__ float ur[DFFN];
  __shared__ float red[DD];
  int tid = threadIdx.x; size_t row = blockIdx.x;
  ur[tid]       = u1[row * DFFN + tid];
  ur[tid + 256] = u1[row * DFFN + tid + 256];
  __syncthreads();
  float acc = b2[tid] + z2[row * DD + tid];
  const float* wr = W2 + tid * DFFN;
  for (int k = 0; k < DFFN; ++k) acc += ur[k] * wr[k];
  uu[row * DD + tid] = block_ln(acc, red, tid, DD);
}

__global__ void k_out(const float* __restrict__ uu, const float* __restrict__ ow,
                      const float* __restrict__ ob, const float* __restrict__ gl,
                      float* __restrict__ y) {
  __shared__ float red[DD];
  int tid = threadIdx.x; size_t row = blockIdx.x;
  red[tid] = uu[row * DD + tid] * ow[tid];
  __syncthreads();
  for (int s = 128; s > 0; s >>= 1) { if (tid < s) red[tid] += red[tid + s]; __syncthreads(); }
  if (tid == 0) {
    int p = (int)(row % PP);
    float val = red[0] + ob[0];
    float l0 = gl[p * 3], l1 = gl[p * 3 + 1], l2 = gl[p * 3 + 2];
    float mx = fmaxf(l0, fmaxf(l1, l2));
    float e0 = __expf(l0 - mx), e1 = __expf(l1 - mx), e2 = __expf(l2 - mx);
    y[row] = (e1 + e2) / (e0 + e1 + e2) * val;
  }
}

extern "C" void kernel_launch(void* const* d_in, const int* in_sizes, int n_in,
                              void* d_out, int out_size, void* d_ws, size_t ws_size,
                              hipStream_t stream) {
  (void)in_sizes; (void)n_in; (void)out_size; (void)ws_size;
  const float* x    = (const float*)d_in[0];
  const float* dww  = (const float*)d_in[2];
  const float* pww  = (const float*)d_in[3];
  const float* bng  = (const float*)d_in[4];
  const float* bnb  = (const float*)d_in[5];
  const float* lgWq = (const float*)d_in[6];
  const float* lgWk = (const float*)d_in[7];
  const float* lgWv = (const float*)d_in[8];
  const float* lgWo = (const float*)d_in[9];
  const float* dQd  = (const float*)d_in[10];
  const float* dWq  = (const float*)d_in[11];
  const float* dWk  = (const float*)d_in[12];
  const float* dWv  = (const float*)d_in[13];
  const float* dWo  = (const float*)d_in[14];
  const float* f1w  = (const float*)d_in[15];
  const float* f1b  = (const float*)d_in[16];
  const float* f2w  = (const float*)d_in[17];
  const float* f2b  = (const float*)d_in[18];
  const float* ow   = (const float*)d_in[19];
  const float* obp  = (const float*)d_in[20];
  const float* gl   = (const float*)d_in[21];

  char* wsp = (char*)d_ws;
  size_t off = 0;
  auto alloc = [&](size_t bytes) -> void* {
    void* p = wsp + off;
    off = (off + bytes + 255) & ~(size_t)255;
    return p;
  };
  u16*   z    = (u16*)  alloc((size_t)BB * SS * DD * 2);
  u16*   Kp   = (u16*)  alloc((size_t)BB * SS * DD * 2);
  u16*   Vp   = (u16*)  alloc((size_t)BB * SS * DD * 2);
  float* Qf   = (float*)alloc((size_t)BB * MM * DD * 4);
  u16*   Qb   = (u16*)  alloc((size_t)BB * MM * DD * 2);
  u16*   Qh   = (u16*)  alloc((size_t)BB * MM * DD * 2);
  float* Obuf = (float*)alloc((size_t)BB * MM * DD * 4);
  float* Hh   = (float*)alloc((size_t)BB * MM * DD * 4);
  float* k2   = (float*)alloc((size_t)BB * MM * DD * 4);
  float* v2   = (float*)alloc((size_t)BB * MM * DD * 4);
  float* q2   = (float*)alloc((size_t)PP * DD * 4);
  float* o2   = (float*)alloc((size_t)BB * PP * DD * 4);
  float* z2   = (float*)alloc((size_t)BB * PP * DD * 4);
  float* u1   = (float*)alloc((size_t)BB * PP * DFFN * 4);
  float* uu   = (float*)alloc((size_t)BB * PP * DD * 4);
  float* st   = (float*)alloc((size_t)4 * DD * 4);
  u16*   pwb  = (u16*)  alloc((size_t)DD * CIN * 2);
  u16*   Wqb  = (u16*)  alloc((size_t)DD * DD * 2);
  u16*   Wkb  = (u16*)  alloc((size_t)DD * DD * 2);
  u16*   Wvb  = (u16*)  alloc((size_t)DD * DD * 2);

  float* y     = (float*)d_out;
  float* attng = y + (size_t)BB * PP;
  float* attnd = attng + (size_t)BB * HH * MM * SS;

  k_zero<<<1, 512, 0, stream>>>(st, 2 * DD);
  k_cvtw<<<(DD * CIN + 255) / 256, 256, 0, stream>>>(pww, pwb, DD * CIN);
  k_cvtw<<<(DD * DD + 255) / 256, 256, 0, stream>>>(lgWq, Wqb, DD * DD);
  k_cvtw<<<(DD * DD + 255) / 256, 256, 0, stream>>>(lgWk, Wkb, DD * DD);
  k_cvtw<<<(DD * DD + 255) / 256, 256, 0, stream>>>(lgWv, Wvb, DD * DD);
  k_frontend<<<BB * (SS / 64), 256, 0, stream>>>(x, dww, pwb, z, st);
  k_bnfin<<<1, DD, 0, stream>>>(st, bng, bnb);
  k_bnapply<<<(BB * SS * DD) / 2048, 256, 0, stream>>>(z, st);
  k_qmean<<<BB * MM, 256, 0, stream>>>(z, Qf, Qb);
  k_proj<<<(BB * SS) / 8, 256, 0, stream>>>(z, Wkb, Kp);
  k_proj<<<(BB * SS) / 8, 256, 0, stream>>>(z, Wvb, Vp);
  k_proj<<<(BB * MM) / 8, 256, 0, stream>>>(Qb, Wqb, Qh);
  size_t smem = (size_t)(16 * SROW + 256 + 32 + 16 * 32) * sizeof(float);
  k_attn<<<BB * HH * MTILES, 256, smem, stream>>>(Qh, Kp, Vp, attng, Obuf);
  k_lgout<<<BB * MM, 256, 0, stream>>>(Obuf, lgWo, Qf, Hh);
  k_decq<<<PP, 256, 0, stream>>>(dQd, dWq, q2);
  k_deckv<<<BB * MM, 256, 0, stream>>>(Hh, dWk, dWv, k2, v2);
  k_decattn<<<BB * HH, 256, 0, stream>>>(q2, k2, v2, attnd, o2);
  k_decz2<<<BB * PP, 256, 0, stream>>>(o2, dWo, dQd, z2);
  k_ffn1<<<BB * PP, 256, 0, stream>>>(z2, f1w, f1b, u1);
  k_ffn2<<<BB * PP, 256, 0, stream>>>(u1, f2w, f2b, z2, uu);
  k_out<<<BB * PP, 256, 0, stream>>>(uu, ow, obp, gl, y);
}
